// SAttention_76785425318727
// MI455X (gfx1250) — compile-verified, matrix operand order checked by reading, not run on hardware
//
#include <hip/hip_runtime.h>
#include <hip/hip_bf16.h>
#include <math.h>

// ---------------------------------------------------------------------------
// Transformer block for MI455X (gfx1250, wave32, WMMA).
// GEMMs: v_wmma_f32_16x16x32_bf16 with LDS-staged tiles fed by CDNA5 async
// global->LDS copies (ASYNCcnt), double-buffered. Attention is flash-style
// with online softmax; LDS used for P-fragment relayout (ds_load_b128).
// ---------------------------------------------------------------------------

#define BDIM   4
#define LSEQ   2048
#define DMODEL 1024
#define NHEAD  16
#define HDIM   64
#define MROWS  (BDIM * LSEQ)   // 8192

typedef __bf16 bf16_t;
typedef bf16_t bf16x8 __attribute__((ext_vector_type(8)));
typedef bf16_t v16bf  __attribute__((ext_vector_type(16)));
typedef float  v8f    __attribute__((ext_vector_type(8)));
typedef int    v4i_t  __attribute__((vector_size(4 * sizeof(int))));

// Load a 16-bit WMMA fragment (A or B role) for one lane from a row-major
// row pointer. 16-bit 16x32 layout: lane half h = lane/16 owns K-chunks
// [h*8, h*8+8) and [16+h*8, 16+h*8+8)  -> two b128 loads (global or ds).
__device__ __forceinline__ v16bf load_frag(const __hip_bfloat16* p, int half) {
  bf16x8 lo = *reinterpret_cast<const bf16x8*>(p + half * 8);
  bf16x8 hi = *reinterpret_cast<const bf16x8*>(p + 16 + half * 8);
  return __builtin_shufflevector(lo, hi, 0, 1, 2, 3, 4, 5, 6, 7,
                                 8, 9, 10, 11, 12, 13, 14, 15);
}

__device__ __forceinline__ v8f wmma_bf16(v16bf a, v16bf b, v8f c) {
  return __builtin_amdgcn_wmma_f32_16x16x32_bf16(false, a, false, b,
                                                 (short)0, c, false, false);
}

// Async global->LDS 16-byte copy (CDNA5, tracked by ASYNCcnt).
// Probe-discovered signature: (global v4i*, local v4i*, imm offset, imm cpol).
typedef __attribute__((address_space(1))) v4i_t* gv4p;
typedef __attribute__((address_space(3))) v4i_t* lv4p;

__device__ __forceinline__ void async_b128(const __hip_bfloat16* g,
                                           __hip_bfloat16* l) {
#if __has_builtin(__builtin_amdgcn_global_load_async_to_lds_b128)
  __builtin_amdgcn_global_load_async_to_lds_b128((gv4p)(void*)(size_t)(const void*)g,
                                                 (lv4p)(void*)l, 0, 0);
#else
  // LDS byte offset = low 32 bits of the generic (flat) address.
  unsigned laddr = (unsigned)(unsigned long long)l;
  asm volatile("global_load_async_to_lds_b128 %0, %1, off"
               :: "v"(laddr), "v"((unsigned long long)g) : "memory");
#endif
}

#if __has_builtin(__builtin_amdgcn_s_wait_asynccnt)
#define WAIT_ASYNC(n) __builtin_amdgcn_s_wait_asynccnt(n)
#else
#define WAIT_ASYNC(n) asm volatile("s_wait_asynccnt %0" ::"i"(n) : "memory")
#endif

// ---------------------------------------------------------------------------
// fp32 [K,N] -> bf16 [N,K] tiled transpose (weights; runs once per launch).
// ---------------------------------------------------------------------------
__global__ __launch_bounds__(256)
void transpose_to_bf16(const float* __restrict__ W, __hip_bfloat16* __restrict__ Wt,
                       int K, int N) {
  __shared__ float tile[32][33];
  int tx = threadIdx.x, ty = threadIdx.y;
  int n0 = blockIdx.x * 32, k0 = blockIdx.y * 32;
  for (int j = ty; j < 32; j += 8)
    tile[j][tx] = W[(size_t)(k0 + j) * N + (n0 + tx)];
  __syncthreads();
  for (int j = ty; j < 32; j += 8)
    Wt[(size_t)(n0 + j) * K + (k0 + tx)] = __float2bfloat16(tile[tx][j]);
}

// ---------------------------------------------------------------------------
// LayerNorm: one 128-thread block per row of 1024. Writes bf16, optional fp32.
// ---------------------------------------------------------------------------
__global__ __launch_bounds__(128)
void layernorm_kernel(const float* __restrict__ x, const float* __restrict__ g,
                      const float* __restrict__ bta, float* __restrict__ y32,
                      __hip_bfloat16* __restrict__ y16) {
  __shared__ float red[128], red2[128];
  const int row = blockIdx.x, t = threadIdx.x;
  const float* xr = x + (size_t)row * DMODEL;
  float s = 0.f, s2 = 0.f;
#pragma unroll
  for (int i = 0; i < 8; i++) {
    float v = xr[t + i * 128];
    s += v; s2 += v * v;
  }
  red[t] = s; red2[t] = s2;
  __syncthreads();
  for (int off = 64; off > 0; off >>= 1) {
    if (t < off) { red[t] += red[t + off]; red2[t] += red2[t + off]; }
    __syncthreads();
  }
  const float mu = red[0] * (1.f / DMODEL);
  const float var = red2[0] * (1.f / DMODEL) - mu * mu;
  const float rstd = rsqrtf(var + 1e-5f);
#pragma unroll
  for (int i = 0; i < 8; i++) {
    int c = t + i * 128;
    float v = (xr[c] - mu) * rstd * g[c] + bta[c];
    y16[(size_t)row * DMODEL + c] = __float2bfloat16(v);
    if (y32) y32[(size_t)row * DMODEL + c] = v;
  }
}

// ---------------------------------------------------------------------------
// bf16 WMMA GEMM: C[m,n] = sum_k A[m,k] * Wt[n,k] + bias[n].
// Block 256 = 8 waves; WG tile 128x128; wave tile 64x32 (8 acc frags).
// A/B tiles (128x32) staged in LDS by async global->LDS b128 copies,
// double-buffered: DMA of tile k+1 overlaps WMMAs on tile k.
// Epilogue modes:
//   0: fp32 out            1: QKV scatter (Q,K [B,H,L,HD] / V [B,H,HD,L])
//   2: exact GELU -> bf16  3: + addf (residual) -> fp32 out
// ---------------------------------------------------------------------------
#define TPAD 40  // 32 + 8 pad; row stride 80B keeps 16B alignment, spreads banks

__global__ __launch_bounds__(256)
void gemm_bf16(const __hip_bfloat16* __restrict__ A,
               const __hip_bfloat16* __restrict__ Wt,
               const float* __restrict__ bias,
               int Nsz, int Ksz, int mode,
               float* __restrict__ outf, __hip_bfloat16* __restrict__ outb,
               const float* __restrict__ addf,
               __hip_bfloat16* __restrict__ Qp, __hip_bfloat16* __restrict__ Kp,
               __hip_bfloat16* __restrict__ Vtp) {
  __shared__ __align__(16) __hip_bfloat16 As[2][128][TPAD];
  __shared__ __align__(16) __hip_bfloat16 Bs[2][128][TPAD];

  const int lane = threadIdx.x & 31;
  const int wave = threadIdx.x >> 5;
  const int half = lane >> 4;
  const int lc   = lane & 15;
  const int mblock = blockIdx.y * 128;
  const int nblock = blockIdx.x * 128;
  const int mwave = (wave >> 2) * 64;   // wave row offset within WG tile
  const int nwave = (wave & 3) * 32;    // wave col offset within WG tile

  // Per-thread async-copy assignment: 512 16B chunks per 128x32 tile,
  // 2 chunks of A + 2 chunks of B per thread.
  const int cid0 = threadIdx.x;          // chunk ids cid0, cid0+256
  const v8f z8 = {0.f, 0.f, 0.f, 0.f, 0.f, 0.f, 0.f, 0.f};
  v8f acc[4][2];
#pragma unroll
  for (int i = 0; i < 4; i++)
#pragma unroll
    for (int j = 0; j < 2; j++) acc[i][j] = z8;

  // Issue one tile-pair's async copies (4 loads/thread).
  auto issue = [&](int buf, int kk) {
#pragma unroll
    for (int t = 0; t < 2; t++) {
      const int cid = cid0 + t * 256;
      const int row = cid >> 2;
      const int c8  = (cid & 3) * 8;
      async_b128(A  + (size_t)(mblock + row) * Ksz + kk + c8, &As[buf][row][c8]);
      async_b128(Wt + (size_t)(nblock + row) * Ksz + kk + c8, &Bs[buf][row][c8]);
    }
  };

  issue(0, 0);
  int buf = 0;
  for (int kk = 0; kk < Ksz; kk += 32) {
    const bool more = (kk + 32) < Ksz;
    if (more) issue(buf ^ 1, kk + 32);   // overlap DMA with compute
    if (more) { WAIT_ASYNC(4); } else { WAIT_ASYNC(0); }
    __syncthreads();

    v16bf af[4];
#pragma unroll
    for (int tm = 0; tm < 4; tm++)
      af[tm] = load_frag(&As[buf][mwave + tm * 16 + lc][0], half);
#pragma unroll
    for (int tn = 0; tn < 2; tn++) {
      v16bf bf_ = load_frag(&Bs[buf][nwave + tn * 16 + lc][0], half);
#pragma unroll
      for (int tm = 0; tm < 4; tm++)
        acc[tm][tn] = wmma_bf16(af[tm], bf_, acc[tm][tn]);
    }
    __syncthreads();  // everyone done reading buf before it is re-filled
    buf ^= 1;
  }

#pragma unroll
  for (int tn = 0; tn < 2; tn++) {
    const int col = nblock + nwave + tn * 16 + lc;
    const float bia = bias[col];
#pragma unroll
    for (int tm = 0; tm < 4; tm++) {
#pragma unroll
      for (int r = 0; r < 8; r++) {
        const int row = mblock + mwave + tm * 16 + r + 8 * half;
        const float v = acc[tm][tn][r] + bia;
        if (mode == 0) {
          outf[(size_t)row * Nsz + col] = v;
        } else if (mode == 1) {
          const int part = col >> 10, rem = col & 1023;
          const int h = rem >> 6, d = rem & 63;
          const int bb = row >> 11, l = row & 2047;
          const __hip_bfloat16 bv = __float2bfloat16(v);
          const size_t bhl = ((size_t)(bb * NHEAD + h)) * LSEQ;
          if (part == 0)       Qp[(bhl + l) * HDIM + d] = bv;
          else if (part == 1)  Kp[(bhl + l) * HDIM + d] = bv;
          else Vtp[(((size_t)(bb * NHEAD + h)) * HDIM + d) * LSEQ + l] = bv;
        } else if (mode == 2) {
          const float gl = 0.5f * v * (1.f + erff(v * 0.70710678118f));
          outb[(size_t)row * Nsz + col] = __float2bfloat16(gl);
        } else {
          outf[(size_t)row * Nsz + col] = v + addf[(size_t)row * Nsz + col];
        }
      }
    }
  }
}

// ---------------------------------------------------------------------------
// Flash attention. Block 128 = 4 waves; each wave owns one 16-row q tile of
// one (b,h). Online softmax with shfl_xor row reductions (the two 16-lane
// halves of the C layout hold disjoint rows, so xor masks 1/2/4/8 suffice).
// P relayout C-frag -> A-frag goes through per-wave LDS (same-wave LDS ops
// are in-order on CDNA5, no barrier needed).
// ---------------------------------------------------------------------------
__global__ __launch_bounds__(128)
void attn_kernel(const __hip_bfloat16* __restrict__ Q,
                 const __hip_bfloat16* __restrict__ Kc,
                 const __hip_bfloat16* __restrict__ Vt,
                 __hip_bfloat16* __restrict__ O) {
  __shared__ __align__(16) __hip_bfloat16 pt[4][16][40];  // 32 + 8 pad
  const int lane = threadIdx.x & 31;
  const int wave = threadIdx.x >> 5;
  const int half = lane >> 4;
  const int lc   = lane & 15;
  const int w    = blockIdx.x * 4 + wave;   // 0 .. B*H*128-1
  const int bh   = w >> 7;
  const int qbase = (w & 127) << 4;

  const __hip_bfloat16* Qb = Q + (size_t)bh * LSEQ * HDIM;
  const __hip_bfloat16* Kb = Kc + (size_t)bh * LSEQ * HDIM;
  const __hip_bfloat16* Vb = Vt + (size_t)bh * HDIM * LSEQ;

  v16bf qa[2];
#pragma unroll
  for (int dc = 0; dc < 2; dc++)
    qa[dc] = load_frag(Qb + (size_t)(qbase + lc) * HDIM + dc * 32, half);

  const v8f z8 = {0.f, 0.f, 0.f, 0.f, 0.f, 0.f, 0.f, 0.f};
  v8f od[4];
  float mrow[8], lrow[8];
#pragma unroll
  for (int dt = 0; dt < 4; dt++) od[dt] = z8;
#pragma unroll
  for (int r = 0; r < 8; r++) { mrow[r] = -1e30f; lrow[r] = 0.f; }

  for (int kv = 0; kv < LSEQ; kv += 32) {
    v8f s0 = z8, s1 = z8;  // scores for kv cols [0,16) and [16,32)
#pragma unroll
    for (int dc = 0; dc < 2; dc++) {
      v16bf k0 = load_frag(Kb + (size_t)(kv + lc) * HDIM + dc * 32, half);
      v16bf k1 = load_frag(Kb + (size_t)(kv + 16 + lc) * HDIM + dc * 32, half);
      s0 = wmma_bf16(qa[dc], k0, s0);
      s1 = wmma_bf16(qa[dc], k1, s1);
    }
#pragma unroll
    for (int r = 0; r < 8; r++) {
      float a = s0[r] * 0.125f;   // 1/sqrt(64)
      float b = s1[r] * 0.125f;
      float mx = fmaxf(a, b);
      mx = fmaxf(mx, __shfl_xor(mx, 1, 32));
      mx = fmaxf(mx, __shfl_xor(mx, 2, 32));
      mx = fmaxf(mx, __shfl_xor(mx, 4, 32));
      mx = fmaxf(mx, __shfl_xor(mx, 8, 32));
      const float mnew = fmaxf(mrow[r], mx);
      const float alpha = __expf(mrow[r] - mnew);
      const float p0 = __expf(a - mnew);
      const float p1 = __expf(b - mnew);
      float rs = p0 + p1;
      rs += __shfl_xor(rs, 1, 32);
      rs += __shfl_xor(rs, 2, 32);
      rs += __shfl_xor(rs, 4, 32);
      rs += __shfl_xor(rs, 8, 32);
      lrow[r] = lrow[r] * alpha + rs;
      mrow[r] = mnew;
#pragma unroll
      for (int dt = 0; dt < 4; dt++) od[dt][r] *= alpha;
      pt[wave][r + 8 * half][lc]      = __float2bfloat16(p0);
      pt[wave][r + 8 * half][lc + 16] = __float2bfloat16(p1);
    }
    // Reload P as an A-fragment [16q x 32kv] from LDS (ds_load_b128 x2).
    v16bf pa = load_frag(&pt[wave][lc][0], half);
#pragma unroll
    for (int dt = 0; dt < 4; dt++) {
      v16bf vb = load_frag(Vb + (size_t)(dt * 16 + lc) * LSEQ + kv, half);
      od[dt] = wmma_bf16(pa, vb, od[dt]);
    }
  }

  const int b = bh >> 4, h = bh & 15;
#pragma unroll
  for (int dt = 0; dt < 4; dt++) {
#pragma unroll
    for (int r = 0; r < 8; r++) {
      const int qrow = qbase + r + 8 * half;
      const float v = od[dt][r] / lrow[r];
      O[((size_t)(b * LSEQ + qrow)) * DMODEL + h * HDIM + dt * 16 + lc] =
          __float2bfloat16(v);
    }
  }
}

// ---------------------------------------------------------------------------
// Host-side orchestration. Workspace layout (bytes), with reuse:
//   G (ffn hidden, 64MB) overlays Q/K/V/Oproj which are dead by then;
//   attention output O overlays Hb (dead after QKV GEMM).
// ---------------------------------------------------------------------------
extern "C" void kernel_launch(void* const* d_in, const int* in_sizes, int n_in,
                              void* d_out, int out_size, void* d_ws, size_t ws_size,
                              hipStream_t stream) {
  (void)in_sizes; (void)n_in; (void)out_size; (void)ws_size;
  const float* x      = (const float*)d_in[0];
  const float* qkv_w  = (const float*)d_in[1];
  const float* qkv_b  = (const float*)d_in[2];
  const float* out_w  = (const float*)d_in[3];
  const float* out_b  = (const float*)d_in[4];
  const float* ln1_g  = (const float*)d_in[5];
  const float* ln1_b  = (const float*)d_in[6];
  const float* ln2_g  = (const float*)d_in[7];
  const float* ln2_b  = (const float*)d_in[8];
  const float* ffn_w1 = (const float*)d_in[9];
  const float* ffn_b1 = (const float*)d_in[10];
  const float* ffn_w2 = (const float*)d_in[11];
  const float* ffn_b2 = (const float*)d_in[12];
  float* out = (float*)d_out;

  char* ws = (char*)d_ws;
  const size_t OFF_WQKV  = 0;                       // 3072x1024 bf16 = 6MB
  const size_t OFF_WOUT  = 6291456;                 // 1024x1024 bf16 = 2MB
  const size_t OFF_W1    = 8388608;                 // 4096x1024 bf16 = 8MB
  const size_t OFF_W2    = 16777216;                // 1024x4096 bf16 = 8MB
  const size_t OFF_HB    = 25165824;                // 8192x1024 bf16 (also O)
  const size_t OFF_Q     = 41943040;                // [B,H,L,HD] bf16 (also G)
  const size_t OFF_K     = 58720256;
  const size_t OFF_V     = 75497472;                // [B,H,HD,L] bf16
  const size_t OFF_OPROJ = 92274688;                // 8192x1024 fp32
  const size_t OFF_OUTF  = 125829120;               // 8192x1024 fp32 (LN2)
  const size_t OFF_OUTB  = 159383552;               // 8192x1024 bf16 (LN2)
  // total ~168MB; G (8192x4096 bf16 = 64MB) overlays [OFF_Q, OFF_Q+64MB)

  __hip_bfloat16* Wqkv = (__hip_bfloat16*)(ws + OFF_WQKV);
  __hip_bfloat16* Wout = (__hip_bfloat16*)(ws + OFF_WOUT);
  __hip_bfloat16* W1t  = (__hip_bfloat16*)(ws + OFF_W1);
  __hip_bfloat16* W2t  = (__hip_bfloat16*)(ws + OFF_W2);
  __hip_bfloat16* Hb   = (__hip_bfloat16*)(ws + OFF_HB);
  __hip_bfloat16* Ob   = Hb;  // reuse
  __hip_bfloat16* Qp   = (__hip_bfloat16*)(ws + OFF_Q);
  __hip_bfloat16* Kp   = (__hip_bfloat16*)(ws + OFF_K);
  __hip_bfloat16* Vtp  = (__hip_bfloat16*)(ws + OFF_V);
  __hip_bfloat16* G    = (__hip_bfloat16*)(ws + OFF_Q);  // reuse
  float* Oproj = (float*)(ws + OFF_OPROJ);
  float* OutF  = (float*)(ws + OFF_OUTF);
  __hip_bfloat16* OutB = (__hip_bfloat16*)(ws + OFF_OUTB);

  const dim3 tb(32, 8);
  // Weight casts + transposes (fp32 [K,N] -> bf16 [N,K]).
  transpose_to_bf16<<<dim3(3072 / 32, 1024 / 32), tb, 0, stream>>>(qkv_w, Wqkv, 1024, 3072);
  transpose_to_bf16<<<dim3(1024 / 32, 1024 / 32), tb, 0, stream>>>(out_w, Wout, 1024, 1024);
  transpose_to_bf16<<<dim3(4096 / 32, 1024 / 32), tb, 0, stream>>>(ffn_w1, W1t, 1024, 4096);
  transpose_to_bf16<<<dim3(1024 / 32, 4096 / 32), tb, 0, stream>>>(ffn_w2, W2t, 4096, 1024);

  // LN1 -> bf16 activations
  layernorm_kernel<<<MROWS, 128, 0, stream>>>(x, ln1_g, ln1_b, nullptr, Hb);

  // QKV GEMM with scatter epilogue (mode 1)
  gemm_bf16<<<dim3(3072 / 128, MROWS / 128), 256, 0, stream>>>(
      Hb, Wqkv, qkv_b, 3072, 1024, 1, nullptr, nullptr, nullptr, Qp, Kp, Vtp);

  // Flash attention -> O bf16 [B,L,D] (overlays Hb)
  attn_kernel<<<BDIM * NHEAD * (LSEQ / 16) / 4, 128, 0, stream>>>(Qp, Kp, Vtp, Ob);

  // Output projection (mode 0) -> fp32
  gemm_bf16<<<dim3(1024 / 128, MROWS / 128), 256, 0, stream>>>(
      Ob, Wout, out_b, 1024, 1024, 0, Oproj, nullptr, nullptr, nullptr, nullptr, nullptr);

  // LN2 -> fp32 (residual source) + bf16 (FFN input)
  layernorm_kernel<<<MROWS, 128, 0, stream>>>(Oproj, ln2_g, ln2_b, OutF, OutB);

  // FFN1 + exact GELU (mode 2) -> bf16 G (overlays Q/K/V/Oproj)
  gemm_bf16<<<dim3(4096 / 128, MROWS / 128), 256, 0, stream>>>(
      OutB, W1t, ffn_b1, 4096, 1024, 2, nullptr, G, nullptr, nullptr, nullptr, nullptr);

  // FFN2 + residual (mode 3) -> d_out fp32
  gemm_bf16<<<dim3(1024 / 128, MROWS / 128), 256, 0, stream>>>(
      G, W2t, ffn_b2, 1024, 4096, 3, out, nullptr, OutF, nullptr, nullptr, nullptr);
}
